// Transformer_50379966382942
// MI455X (gfx1250) — matrix operand order repaired and checked by reading, not compiled, not run on hardware
//
#include <hip/hip_runtime.h>
#include <hip/hip_bf16.h>

// ---------------- model constants (must match reference) ----------------
#define Dm    512
#define Hh    8
#define DKk   64          // Dm / Hh
#define Bb    4
#define Tt    512
#define NLAYER 6
#define HFD   (4 * Dm)    // 2048
#define MROWS (Bb * Tt)   // 2048
#define VTGT  32000
#define ATT_SCALE 0.125f  // 1/sqrt(64)

#define BT_STRIDE 48      // u16 elements per LDS tile row (96B: 16B aligned, bank-spread)

typedef unsigned short u16;
typedef __attribute__((ext_vector_type(16))) __bf16 v16bf;
typedef __attribute__((ext_vector_type(8)))  __bf16 v8bf;
typedef __attribute__((ext_vector_type(8)))  float  v8f;

union V16 { v16bf v; v8bf h[2]; u16 u[16]; };

__device__ __forceinline__ u16 f2bf(float x) {
    union { float f; unsigned int u; } cv; cv.f = x;
    unsigned int b = cv.u;
    b += 0x7FFFu + ((b >> 16) & 1u);   // round-to-nearest-even
    return (u16)(b >> 16);
}

__device__ __forceinline__ v8f wmma_bf16(v16bf a, v16bf b, v8f c) {
    return __builtin_amdgcn_wmma_f32_16x16x32_bf16(false, a, false, b, (short)0, c,
                                                   false, false);
}

// A fragment (16x32 bf16) from bf16 row-major, leading dim lda (elements).
// Lane m = lane&15; lanes0-15 K={0..15}, lanes16-31 K={16..31}.
__device__ __forceinline__ v16bf load_a_bf(const u16* __restrict__ A, int lda,
                                           int row0, int k0, int lane) {
    V16 r;
    const u16* ap = A + (size_t)(row0 + (lane & 15)) * lda + k0 + ((lane >> 4) << 4);
    r.h[0] = *(const v8bf*)(ap);        // k = kh+0 .. kh+7
    r.h[1] = *(const v8bf*)(ap + 8);    // k = kh+8 .. kh+15
    return r.v;
}

// B fragment (32x16 bf16) from an n-major (transposed) bf16 matrix in global, ld = ldb.
__device__ __forceinline__ v16bf load_b_bfT(const u16* __restrict__ BT, int ldb,
                                            int col0, int k0, int lane) {
    V16 r;
    const u16* bp = BT + (size_t)(col0 + (lane & 15)) * ldb + k0 + ((lane >> 4) << 4);
    r.h[0] = *(const v8bf*)(bp);
    r.h[1] = *(const v8bf*)(bp + 8);
    return r.v;
}

// B fragment from the LDS-staged 32x32 tile (n-major rows, BT_STRIDE u16 apart).
__device__ __forceinline__ v16bf load_b_lds(const u16* bt, int col0, int lane) {
    V16 r;
    const u16* bp = bt + (col0 + (lane & 15)) * BT_STRIDE + ((lane >> 4) << 4);
    r.h[0] = *(const v8bf*)(bp);        // ds_load_b128
    r.h[1] = *(const v8bf*)(bp + 8);
    return r.v;
}

// Cooperative async copy of one 32(n) x 32(k) bf16 tile of WT into LDS.
// 256 threads x 8 bytes each; tracked with ASYNCcnt (CDNA5 async global->LDS path).
__device__ __forceinline__ void async_copy_btile(u16* dst, const u16* __restrict__ WT,
                                                 int ldb, int tn, int k0, int tid) {
    int nrow = tid >> 3, ch = tid & 7;  // 32 rows x 8 chunks of 8B
    unsigned lds = (unsigned)(size_t)(void*)(dst + nrow * BT_STRIDE + ch * 4);
    unsigned long long gp =
        (unsigned long long)(size_t)(WT + (size_t)(tn + nrow) * ldb + k0 + ch * 4);
    asm volatile("global_load_async_to_lds_b64 %0, %1, off" :: "v"(lds), "v"(gp) : "memory");
}

__device__ __forceinline__ void wait_async0() {
    asm volatile("s_wait_asynccnt 0x0" ::: "memory");
}

// Epilogue for one 16x16 accumulator.
// outMode: 0 = fp32 C[M,N] (+resid); 1 = bf16 Cb[M,N]; 2 = bf16 transposed Cb[N,M]
__device__ __forceinline__ void store_tile(v8f acc, int rowb, int col, int M, int N,
                                           int act, const float* __restrict__ bias,
                                           const float* __restrict__ resid,
                                           float* __restrict__ C,
                                           u16* __restrict__ Cb, int outMode) {
    float bv = bias ? bias[col] : 0.f;
#pragma unroll
    for (int i = 0; i < 8; ++i) {
        float v = acc[i] + bv;
        if (act == 1)      v = v > 0.f ? v : (__expf(v) - 1.f);   // ELU
        else if (act == 2) v = fmaxf(v, 0.f);                     // ReLU
        size_t off = (size_t)(rowb + i) * N + col;
        if (resid) v += resid[off];
        if (outMode == 0)      C[off] = v;
        else if (outMode == 1) Cb[off] = f2bf(v);
        else                   Cb[(size_t)col * M + (rowb + i)] = f2bf(v);
    }
}

// ---------------- kernels ----------------

// C = act(A[M,K](bf16) @ W stored n-major (bf16) + bias) (+resid)
// Block = 8 waves stacked along M (256 rows) sharing one 32-wide N tile whose
// weight tile is double-buffered in LDS via async global->LDS copies.
// Requires M % 256 == 0 (all call sites use M = 2048).
__global__ __launch_bounds__(256) void gemm_bf16_kernel(
    const u16* __restrict__ A, const u16* __restrict__ WT,
    const float* __restrict__ bias, const float* __restrict__ resid,
    float* __restrict__ C, u16* __restrict__ Cb,
    int M, int N, int K, int act, int outMode) {
    __shared__ __align__(16) u16 btile[2][32 * BT_STRIDE];
    int tid  = threadIdx.x;
    int lane = tid & 31;
    int wave = tid >> 5;
    int tilesN = N >> 5;
    int bm = blockIdx.x / tilesN;
    int tn = (blockIdx.x % tilesN) << 5;
    int tm = (bm << 8) + (wave << 5);

    async_copy_btile(btile[0], WT, K, tn, 0, tid);

    v8f c00 = {}, c01 = {}, c10 = {}, c11 = {};
    for (int k0 = 0; k0 < K; k0 += 32) {
        int cur = (k0 >> 5) & 1;
        wait_async0();
        __syncthreads();
        if (k0 + 32 < K) {
            async_copy_btile(btile[cur ^ 1], WT, K, tn, k0 + 32, tid);
            __builtin_prefetch(A + (size_t)(tm + (lane & 15)) * K + k0 + 32, 0, 3);
        }
        v16bf a0 = load_a_bf(A, K, tm,      k0, lane);
        v16bf a1 = load_a_bf(A, K, tm + 16, k0, lane);
        v16bf b0 = load_b_lds(btile[cur], 0,  lane);
        v16bf b1 = load_b_lds(btile[cur], 16, lane);
        c00 = wmma_bf16(a0, b0, c00);
        c01 = wmma_bf16(a0, b1, c01);
        c10 = wmma_bf16(a1, b0, c10);
        c11 = wmma_bf16(a1, b1, c11);
        __syncthreads();
    }
    int cl = lane & 15, rb = (lane >> 4) << 3;
    store_tile(c00, tm + rb,      tn + cl,      M, N, act, bias, resid, C, Cb, outMode);
    store_tile(c01, tm + rb,      tn + 16 + cl, M, N, act, bias, resid, C, Cb, outMode);
    store_tile(c10, tm + 16 + rb, tn + cl,      M, N, act, bias, resid, C, Cb, outMode);
    store_tile(c11, tm + 16 + rb, tn + 16 + cl, M, N, act, bias, resid, C, Cb, outMode);
}

// S[z,q,k] = scale * Q . K  per head slice; Q/K are bf16 [B*T, D] row-major.
__global__ __launch_bounds__(256) void attn_scores_kernel(
    const u16* __restrict__ Q, const u16* __restrict__ Kb,
    float* __restrict__ S, float scale) {
    int lane = threadIdx.x & 31;
    int wv   = blockIdx.x * (blockDim.x >> 5) + (threadIdx.x >> 5);
    const int tilesN = Tt >> 5;           // 16
    int tm = (wv / tilesN) << 5, tn = (wv % tilesN) << 5;
    int z = blockIdx.y, b = z / Hh, h = z % Hh;
    const u16* Qp = Q  + (size_t)b * Tt * Dm + h * DKk;   // lda = Dm
    const u16* Kp = Kb + (size_t)b * Tt * Dm + h * DKk;   // n-major view, ld = Dm
    float* Sp = S + (size_t)z * Tt * Tt;

    v8f c00 = {}, c01 = {}, c10 = {}, c11 = {};
#pragma unroll
    for (int k0 = 0; k0 < DKk; k0 += 32) {
        v16bf a0 = load_a_bf(Qp, Dm, tm,      k0, lane);
        v16bf a1 = load_a_bf(Qp, Dm, tm + 16, k0, lane);
        v16bf b0 = load_b_bfT(Kp, Dm, tn,      k0, lane);
        v16bf b1 = load_b_bfT(Kp, Dm, tn + 16, k0, lane);
        c00 = wmma_bf16(a0, b0, c00);
        c01 = wmma_bf16(a0, b1, c01);
        c10 = wmma_bf16(a1, b0, c10);
        c11 = wmma_bf16(a1, b1, c11);
    }
    int cl = lane & 15, rb = (lane >> 4) << 3;
#pragma unroll
    for (int i = 0; i < 8; ++i) {
        Sp[(size_t)(tm + rb + i) * Tt      + tn + cl]      = c00[i] * scale;
        Sp[(size_t)(tm + rb + i) * Tt      + tn + 16 + cl] = c01[i] * scale;
        Sp[(size_t)(tm + 16 + rb + i) * Tt + tn + cl]      = c10[i] * scale;
        Sp[(size_t)(tm + 16 + rb + i) * Tt + tn + 16 + cl] = c11[i] * scale;
    }
}

// Row softmax (pad + optional causal mask) fp32 -> bf16 probabilities.
__global__ __launch_bounds__(128) void softmax_kernel(
    const float* __restrict__ S, u16* __restrict__ P,
    const int* __restrict__ toks, int causal) {
    __shared__ float red[128];
    int r = blockIdx.x;                      // r = z*Tt + qi
    int z = r / Tt, qi = r % Tt, b = z / Hh;
    const float* row = S + (size_t)r * Tt;
    u16* prow = P + (size_t)r * Tt;
    const int* tb = toks + b * Tt;
    int tid = threadIdx.x;

    float v[Tt / 128];
    float mx = -1e30f;
#pragma unroll
    for (int c = 0; c < Tt / 128; ++c) {
        int j = tid + c * 128;
        bool msk = (tb[j] == 0) || (causal && j > qi);
        v[c] = msk ? -1e30f : row[j];
        mx = fmaxf(mx, v[c]);
    }
    red[tid] = mx; __syncthreads();
    for (int s = 64; s > 0; s >>= 1) { if (tid < s) red[tid] = fmaxf(red[tid], red[tid + s]); __syncthreads(); }
    mx = red[0]; __syncthreads();

    float sum = 0.f;
#pragma unroll
    for (int c = 0; c < Tt / 128; ++c) {
        float e = (v[c] <= -1e29f) ? 0.f : __expf(v[c] - mx);
        v[c] = e; sum += e;
    }
    red[tid] = sum; __syncthreads();
    for (int s = 64; s > 0; s >>= 1) { if (tid < s) red[tid] += red[tid + s]; __syncthreads(); }
    float inv = 1.f / red[0];
#pragma unroll
    for (int c = 0; c < Tt / 128; ++c) prow[tid + c * 128] = f2bf(v[c] * inv);
}

// O[b,q,h*DK+d] = sum_k P[z,q,k] * V[b,k,h*DK+d]; P bf16 [z][T][T], V transposed bf16 [D][B*T].
__global__ __launch_bounds__(256) void attn_out_kernel(
    const u16* __restrict__ P, const u16* __restrict__ VT, u16* __restrict__ Ob) {
    int lane = threadIdx.x & 31;
    int wv   = blockIdx.x * (blockDim.x >> 5) + (threadIdx.x >> 5);
    const int tilesN = DKk >> 5;              // 2
    int tm = (wv / tilesN) << 5, tn = (wv % tilesN) << 5;
    int z = blockIdx.y, b = z / Hh, h = z % Hh;
    const u16* Pp = P  + (size_t)z * Tt * Tt;                        // lda = Tt
    const u16* Vp = VT + (size_t)(h * DKk) * MROWS + (size_t)b * Tt; // ld = MROWS

    v8f c00 = {}, c01 = {}, c10 = {}, c11 = {};
    for (int k0 = 0; k0 < Tt; k0 += 32) {
        v16bf a0 = load_a_bf(Pp, Tt, tm,      k0, lane);
        v16bf a1 = load_a_bf(Pp, Tt, tm + 16, k0, lane);
        v16bf b0 = load_b_bfT(Vp, MROWS, tn,      k0, lane);
        v16bf b1 = load_b_bfT(Vp, MROWS, tn + 16, k0, lane);
        c00 = wmma_bf16(a0, b0, c00);
        c01 = wmma_bf16(a0, b1, c01);
        c10 = wmma_bf16(a1, b0, c10);
        c11 = wmma_bf16(a1, b1, c11);
    }
    int cl = lane & 15, rb = (lane >> 4) << 3;
#pragma unroll
    for (int i = 0; i < 8; ++i) {
        Ob[(size_t)(b * Tt + tm + rb + i) * Dm      + h * DKk + tn + cl]      = f2bf(c00[i]);
        Ob[(size_t)(b * Tt + tm + rb + i) * Dm      + h * DKk + tn + 16 + cl] = f2bf(c01[i]);
        Ob[(size_t)(b * Tt + tm + 16 + rb + i) * Dm + h * DKk + tn + cl]      = f2bf(c10[i]);
        Ob[(size_t)(b * Tt + tm + 16 + rb + i) * Dm + h * DKk + tn + 16 + cl] = f2bf(c11[i]);
    }
}

__global__ void embed_kernel(const int* __restrict__ toks,
                             const float* __restrict__ we,
                             const float* __restrict__ pe,
                             float* __restrict__ out, int total) {
    int i = blockIdx.x * blockDim.x + threadIdx.x;
    if (i >= total) return;
    int d  = i & (Dm - 1);
    int bt = i >> 9;              // Dm == 512
    int t  = bt & (Tt - 1);
    out[i] = we[(size_t)toks[bt] * Dm + d] + pe[(size_t)t * Dm + d];
}

// LayerNorm fp32 -> bf16 (LN outputs only ever feed WMMA GEMMs).
__global__ __launch_bounds__(128) void layernorm_kernel(
    const float* __restrict__ X, const float* __restrict__ g,
    const float* __restrict__ bt, u16* __restrict__ Yb) {
    __shared__ float red[128];
    int r = blockIdx.x, tid = threadIdx.x;
    const float* x = X + (size_t)r * Dm;
    u16* y = Yb + (size_t)r * Dm;

    float s = 0.f;
    for (int j = tid; j < Dm; j += 128) s += x[j];
    red[tid] = s; __syncthreads();
    for (int st = 64; st > 0; st >>= 1) { if (tid < st) red[tid] += red[tid + st]; __syncthreads(); }
    float mu = red[0] / (float)Dm; __syncthreads();

    float vs = 0.f;
    for (int j = tid; j < Dm; j += 128) { float d = x[j] - mu; vs += d * d; }
    red[tid] = vs; __syncthreads();
    for (int st = 64; st > 0; st >>= 1) { if (tid < st) red[tid] += red[tid + st]; __syncthreads(); }
    float inv = rsqrtf(red[0] / (float)Dm + 1e-5f);

    for (int j = tid; j < Dm; j += 128)
        y[j] = f2bf((x[j] - mu) * inv * g[j] + bt[j]);
}

// Weight convert + transpose: WT[n*K + k] = bf16(W[k*N + n]).
__global__ void cvt_wT_kernel(const float* __restrict__ W, u16* __restrict__ WT,
                              int K, int N) {
    int i = blockIdx.x * blockDim.x + threadIdx.x;
    if (i >= K * N) return;
    int k = i / N, n = i % N;
    WT[(size_t)n * K + k] = f2bf(W[i]);
}

// ---------------- host orchestration ----------------
extern "C" void kernel_launch(void* const* d_in, const int* in_sizes, int n_in,
                              void* d_out, int out_size, void* d_ws, size_t ws_size,
                              hipStream_t stream) {
    const int* Xs = (const int*)d_in[0];
    const int* Xt = (const int*)d_in[1];
    const float* enc_we = (const float*)d_in[2];
    const float* enc_pe = (const float*)d_in[3];
    const float* dec_we = (const float*)d_in[4];
    const float* dec_pe = (const float*)d_in[5];
    auto F = [&](int i) { return (const float*)d_in[i]; };
    const int ENC0 = 6, DEC0 = 78, ENC_LN = 186, LAST_LN = 188, WOUT = 190;

    // fp32 region
    float* fp = (float*)d_ws;
    auto allocF = [&](size_t n) { float* r = fp; fp += n; return r; };
    const size_t ND = (size_t)MROWS * Dm;
    float* x  = allocF(ND);
    float* x2 = allocF(ND);
    float* y  = allocF(ND);
    float* y2 = allocF(ND);
    float* sc = allocF((size_t)Bb * Hh * Tt * Tt);
    // bf16 region
    u16* up = (u16*)fp;
    auto allocU = [&](size_t n) { u16* r = up; up += n; return r; };
    u16* tln  = allocU(ND);                        // LN output
    u16* qb   = allocU(ND);
    u16* kb   = allocU(ND);
    u16* vT   = allocU(ND);                        // V projection, transposed [D][B*T]
    u16* ao   = allocU(ND);                        // attention output
    u16* encb = allocU(ND);                        // encoder final LN
    u16* hb   = allocU((size_t)MROWS * HFD);       // MLP hidden
    u16* pb   = allocU((size_t)Bb * Hh * Tt * Tt); // softmax probs

    // transposed bf16 weights, indexed by flattened input index
    u16* wbf[191] = {};
    auto cvtw = [&](int idx, int K, int N) {
        u16* w = allocU((size_t)K * N);
        wbf[idx] = w;
        int tot = K * N;
        cvt_wT_kernel<<<(tot + 255) / 256, 256, 0, stream>>>(F(idx), w, K, N);
    };
    for (int l = 0; l < NLAYER; ++l) {
        int e = ENC0 + 12 * l;
        for (int j = 2; j <= 5; ++j) cvtw(e + j, Dm, Dm);
        cvtw(e + 8, Dm, HFD);
        cvtw(e + 10, HFD, Dm);
        int d0 = DEC0 + 18 * l;
        for (int j = 2; j <= 5; ++j) cvtw(d0 + j, Dm, Dm);
        for (int j = 8; j <= 11; ++j) cvtw(d0 + j, Dm, Dm);
        cvtw(d0 + 14, Dm, HFD);
        cvtw(d0 + 16, HFD, Dm);
    }
    cvtw(WOUT, Dm, VTGT);

    auto gemm = [&](const u16* A, const u16* WT, const float* bias, const float* resid,
                    float* C, u16* Cb, int M, int N, int K, int act, int outMode) {
        int blocks = (M >> 8) * (N >> 5);   // M % 256 == 0 at every call site
        gemm_bf16_kernel<<<dim3(blocks), dim3(256), 0, stream>>>(
            A, WT, bias, resid, C, Cb, M, N, K, act, outMode);
    };
    auto lnorm = [&](const float* X, int gi, u16* Y) {
        layernorm_kernel<<<MROWS, 128, 0, stream>>>(X, F(gi), F(gi + 1), Y);
    };
    auto attention = [&](const u16* xq, const u16* xkv, int bse,
                         const int* toks, int causal, const float* xres, float* xdst) {
        gemm(xq,  wbf[bse + 0], nullptr, nullptr, nullptr, qb, MROWS, Dm, Dm, 0, 1);
        gemm(xkv, wbf[bse + 1], nullptr, nullptr, nullptr, kb, MROWS, Dm, Dm, 0, 1);
        gemm(xkv, wbf[bse + 2], nullptr, nullptr, nullptr, vT, MROWS, Dm, Dm, 0, 2);
        attn_scores_kernel<<<dim3(256 / 8, Bb * Hh), dim3(256), 0, stream>>>(qb, kb, sc, ATT_SCALE);
        softmax_kernel<<<Bb * Hh * Tt, 128, 0, stream>>>(sc, pb, toks, causal);
        attn_out_kernel<<<dim3(32 / 8, Bb * Hh), dim3(256), 0, stream>>>(pb, vT, ao);
        gemm(ao, wbf[bse + 3], nullptr, xres, xdst, nullptr, MROWS, Dm, Dm, 0, 0);
    };

    // ---------------- encoder ----------------
    embed_kernel<<<(MROWS * Dm + 255) / 256, 256, 0, stream>>>(Xs, enc_we, enc_pe, x, MROWS * Dm);
    for (int l = 0; l < NLAYER; ++l) {
        int e = ENC0 + 12 * l;
        lnorm(x, e + 0, tln);
        attention(tln, tln, e + 2, Xs, /*causal=*/0, x, x2);
        { float* t = x; x = x2; x2 = t; }
        lnorm(x, e + 6, tln);
        gemm(tln, wbf[e + 8], F(e + 9), nullptr, nullptr, hb, MROWS, HFD, Dm, /*ELU*/1, 1);
        gemm(hb, wbf[e + 10], F(e + 11), x, x2, nullptr, MROWS, Dm, HFD, 0, 0);
        { float* t = x; x = x2; x2 = t; }
    }
    lnorm(x, ENC_LN, encb);

    // ---------------- decoder ----------------
    embed_kernel<<<(MROWS * Dm + 255) / 256, 256, 0, stream>>>(Xt, dec_we, dec_pe, y, MROWS * Dm);
    for (int l = 0; l < NLAYER; ++l) {
        int d0 = DEC0 + 18 * l;
        lnorm(y, d0 + 0, tln);
        attention(tln, tln, d0 + 2, Xt, /*causal=*/1, y, y2);
        { float* t = y; y = y2; y2 = t; }
        lnorm(y, d0 + 6, tln);
        attention(tln, encb, d0 + 8, Xs, /*causal=*/0, y, y2);
        { float* t = y; y = y2; y2 = t; }
        lnorm(y, d0 + 12, tln);
        gemm(tln, wbf[d0 + 14], F(d0 + 15), nullptr, nullptr, hb, MROWS, HFD, Dm, /*ReLU*/2, 1);
        gemm(hb, wbf[d0 + 16], F(d0 + 17), y, y2, nullptr, MROWS, Dm, HFD, 0, 0);
        { float* t = y; y = y2; y2 = t; }
    }
    lnorm(y, LAST_LN, tln);
    gemm(tln, wbf[WOUT], nullptr, nullptr, (float*)d_out, nullptr, MROWS, VTGT, Dm, 0, 0);
}
